// ESN_2D_9775345566110
// MI455X (gfx1250) — matrix operand order repaired and checked by reading, not compiled, not run on hardware
//
#include <hip/hip_runtime.h>
#include <hip/hip_bf16.h>

// ---------------- problem constants ----------------
#define B_    64
#define H_    64
#define W_    64
#define R_    128
#define HP    65            // halo height  (index h+1)
#define WP    65            // halo width   (index w+1)
#define CELLH (B_ * R_)     // halves per state cell = 8192
#define D_    256
#define LL    (63 * 63)     // 3969
#define ALPHA 0.9f

typedef _Float16 h8  __attribute__((ext_vector_type(8)));
typedef _Float16 h16 __attribute__((ext_vector_type(16)));
typedef float    f8  __attribute__((ext_vector_type(8)));

__device__ __forceinline__ h16 cat8(h8 lo, h8 hi) {
    return __builtin_shufflevector(lo, hi, 0,1,2,3,4,5,6,7,8,9,10,11,12,13,14,15);
}
__device__ __forceinline__ f8 wmma_f16(h16 a, h16 b, f8 c) {
    // D = A(16x32,f16) * B(32x16,f16) + C(16x16,f32)
    return __builtin_amdgcn_wmma_f32_16x16x32_f16(false, a, false, b, (short)0, c, false, false);
}

// A-operand (16x32 f16, row-major source): lane m=lane&15 holds row m,
// halves K = {hs*8..hs*8+7, 16+hs*8..16+hs*8+7}, hs = lane>>4.
__device__ __forceinline__ h16 load_a_rowmajor(const _Float16* rowbase, int hs) {
    h8 lo = *(const h8*)(rowbase + hs * 8);
    h8 hi = *(const h8*)(rowbase + 16 + hs * 8);
    return cat8(lo, hi);
}
// B-operand (32x16 f16) from K-major (transposed) storage: lane n=lane&15 holds
// column n, K = {hs*8..hs*8+7, 16+hs*8..16+hs*8+7}, hs = lane>>4.
__device__ __forceinline__ h16 load_b_kmajor(const _Float16* colbase, int lane) {
    const _Float16* p = colbase + (lane >> 4) * 8;
    h8 lo = *(const h8*)(p);
    h8 hi = *(const h8*)(p + 16);
    return cat8(lo, hi);
}

// ---------------- init kernels ----------------
__global__ void k_zero(uint4* __restrict__ p, size_t n16) {
    size_t idx    = (size_t)blockIdx.x * blockDim.x + threadIdx.x;
    size_t stride = (size_t)gridDim.x * blockDim.x;
    uint4  z      = {0u, 0u, 0u, 0u};
    for (size_t i = idx; i < n16; i += stride) p[i] = z;
}

__global__ void k_prep_weights(const float* __restrict__ w1, const float* __restrict__ w2,
                               _Float16* __restrict__ w1t, _Float16* __restrict__ w2t) {
    int idx = blockIdx.x * 256 + threadIdx.x;
    if (idx < R_ * R_) {
        int k = idx >> 7, n = idx & 127;           // w[k][n] -> wt[n][k]
        w1t[n * R_ + k] = (_Float16)w1[k * R_ + n];
        w2t[n * R_ + k] = (_Float16)w2[k * R_ + n];
    }
}

// ---------------- persistent wavefront recurrence ----------------
// One workgroup per COLUMN w (64 WGs). WG iterates h = 0..63. "Above" state
// lives in LDS (program order); "left" state is read from global, guarded by
// per-column progress flags (acquire/release, agent scope). WG w only waits on
// WG w-1 -> acyclic -> deadlock-free regardless of residency.
// Wave wv owns N-tile nt=wv; weight B-tiles live in registers for the whole run.
__global__ void k_recur(const float* __restrict__ x, const float* __restrict__ Win,
                        const _Float16* __restrict__ w1t, const _Float16* __restrict__ w2t,
                        _Float16* __restrict__ state, int* __restrict__ flags) {
    __shared__ __align__(16) _Float16 sState[CELLH];  // above-row state, 16 KB
    __shared__ float xs[B_];                          // input column x(h,w,:) per row

    const int w    = blockIdx.x;
    const int tid  = threadIdx.x;
    const int wv   = tid >> 5;       // wave = N-tile 0..7
    const int lane = tid & 31;
    const int nt   = wv;
    const int ln   = lane & 15;
    const int hs   = lane >> 4;
    const int rc   = nt * 16 + ln;   // reservoir output index 0..127

    // zero the "above" buffer for h = 0
    for (int q = tid; q < CELLH / 2; q += 256) ((unsigned int*)sState)[q] = 0u;

    // preload this wave's weight B-tiles into registers (64 VGPRs, held)
    h16 Bw1[4], Bw2[4];
#pragma unroll
    for (int kc = 0; kc < 4; ++kc) {
        Bw1[kc] = load_b_kmajor(w1t + (nt * 16 + ln) * R_ + kc * 32, lane);
        Bw2[kc] = load_b_kmajor(w2t + (nt * 16 + ln) * R_ + kc * 32, lane);
    }
    const float win = Win[rc];

    int* flagL = flags + (w - 1);
    __syncthreads();   // sState zeroed

    for (int h = 0; h < H_; ++h) {
        if (w > 0) {   // wait for left neighbor cell (h, w-1); every thread acquires
            while (__hip_atomic_load(flagL, __ATOMIC_ACQUIRE, __HIP_MEMORY_SCOPE_AGENT) < h + 1)
                __builtin_amdgcn_s_sleep(1);
        }
        if (tid < B_) xs[tid] = x[(size_t)tid * (H_ * W_) + h * W_ + w];

        const _Float16* Sl = state + (size_t)((h + 1) * WP + w) * CELLH;        // left (halo)
        _Float16*       Sc = state + (size_t)((h + 1) * WP + (w + 1)) * CELLH;  // current

        f8 accs[4];
#pragma unroll
        for (int mt = 0; mt < 4; ++mt) {
            f8 acc = {};
#pragma unroll
            for (int kc = 0; kc < 4; ++kc) {
                h16 aL = load_a_rowmajor(Sl + (mt * 16 + ln) * R_ + kc * 32, hs);
                h16 aA = load_a_rowmajor(&sState[(mt * 16 + ln) * R_ + kc * 32], hs);
                acc = wmma_f16(aL, Bw1[kc], acc);
                acc = wmma_f16(aA, Bw2[kc], acc);
            }
            accs[mt] = acc;
        }
        __syncthreads();   // all reads of sState/xs done -> safe to overwrite

#pragma unroll
        for (int mt = 0; mt < 4; ++mt) {
#pragma unroll
            for (int i = 0; i < 8; ++i) {
                const int bI = mt * 16 + i + hs * 8;   // batch index
                float v = accs[mt][i] + xs[bI] * win;
                v = tanhf(v);
                const _Float16 hv = (_Float16)v;
                sState[bI * R_ + rc] = hv;             // "above" for next row
                Sc[bI * R_ + rc]     = hv;             // for right neighbor + Gram
            }
        }
        __threadfence();   // each thread: global stores visible at agent scope
        __syncthreads();   // whole WG done with row h
        if (tid == 0)
            __hip_atomic_store(flags + w, h + 1, __ATOMIC_RELEASE, __HIP_MEMORY_SCOPE_AGENT);
    }
}

// ---------------- Gram matrix XtX = pre^T * pre (per batch) ----------------
// grid (64 batches, 4 row-groups); each WG: 4 d1-tiles x 16 d2-tiles, 8 tiles/wave.
__global__ void k_gram(const _Float16* __restrict__ state, float* __restrict__ XtX) {
    __shared__ __align__(16) _Float16 sT[D_ * 32];   // transposed chunk: sT[d][k], 16 KB

    const int b   = blockIdx.x;
    const int g   = blockIdx.y;
    const int tid = threadIdx.x;
    const int wv  = tid >> 5;
    const int lane = tid & 31;
    const int d1t = g * 4 + (wv >> 1);
    const int ngrp = wv & 1;
    const int ln  = lane & 15;
    const int hs  = lane >> 4;

    const int lr  = tid >> 3;   // staging: chunk row 0..31
    const int seg = tid & 7;    // staging: 32-half segment (0-3 left, 4-7 above)

    f8 acc[8] = {};

    const int nchunk = (LL + 31) / 32;   // 125 (last chunk zero-padded)
    for (int kc = 0; kc < nchunk; ++kc) {
        const int l = kc * 32 + lr;
        _Float16 v[32];
        if (l < LL) {
            const int i = l / 63, j = l - i * 63;
            const _Float16* src;
            if (seg < 4)    // left part: state[b, h=i+1, w=j]   -> halo (i+2, j+1)
                src = state + (size_t)((i + 2) * WP + (j + 1)) * CELLH + b * R_ + seg * 32;
            else            // above part: state[b, h=i, w=j+1]  -> halo (i+1, j+2)
                src = state + (size_t)((i + 1) * WP + (j + 2)) * CELLH + b * R_ + (seg - 4) * 32;
            h8 p0 = *(const h8*)(src);
            h8 p1 = *(const h8*)(src + 8);
            h8 p2 = *(const h8*)(src + 16);
            h8 p3 = *(const h8*)(src + 24);
#pragma unroll
            for (int q = 0; q < 8; ++q) { v[q] = p0[q]; v[8+q] = p1[q]; v[16+q] = p2[q]; v[24+q] = p3[q]; }
        } else {
#pragma unroll
            for (int q = 0; q < 32; ++q) v[q] = (_Float16)0.0f;
        }
        __syncthreads();                         // previous chunk fully consumed
#pragma unroll
        for (int dd = 0; dd < 32; ++dd)
            sT[(seg * 32 + dd) * 32 + lr] = v[dd];  // transpose into LDS
        __syncthreads();

        h16 a = load_a_rowmajor(&sT[(d1t * 16 + ln) * 32], hs);
#pragma unroll
        for (int t = 0; t < 8; ++t) {
            const int d2t = ngrp * 8 + t;
            h16 bb = load_b_kmajor(&sT[(d2t * 16 + ln) * 32], lane);
            acc[t] = wmma_f16(a, bb, acc[t]);
        }
    }

#pragma unroll
    for (int t = 0; t < 8; ++t) {
        const int d2t = ngrp * 8 + t;
        const int col = d2t * 16 + ln;
#pragma unroll
        for (int i = 0; i < 8; ++i) {
            const int row = d1t * 16 + i + hs * 8;
            XtX[(size_t)b * (D_ * D_) + row * D_ + col] = acc[t][i];
        }
    }
}

// ---------------- Xty = pre^T * tgt (per batch) ----------------
__global__ void k_xty(const _Float16* __restrict__ state, const float* __restrict__ x,
                      float* __restrict__ Xty) {
    const int b = blockIdx.x, t = threadIdx.x;
    float acc = 0.f;
    int i = 0, j = 0;
    for (int l = 0; l < LL; ++l) {
        size_t cell; int r;
        if (t < 128) { cell = (size_t)((i + 2) * WP + (j + 1)); r = t; }
        else         { cell = (size_t)((i + 1) * WP + (j + 2)); r = t - 128; }
        const float pv = (float)state[cell * CELLH + b * R_ + r];
        const float tg = x[(size_t)b * (H_ * W_) + (i + 1) * W_ + (j + 1)];
        acc += pv * tg;
        if (++j == 63) { j = 0; ++i; }
    }
    Xty[b * D_ + t] = acc;
}

// ---------------- ridge solve: (XtX + a*I) x = Xty, per batch ----------------
__global__ void k_solve(float* __restrict__ A, const float* __restrict__ Xty,
                        float* __restrict__ out) {
    __shared__ float colk[D_], ys[D_], xs[D_];
    const int b = blockIdx.x, t = threadIdx.x;
    float* Ab = A + (size_t)b * (D_ * D_);

    Ab[t * D_ + t] += ALPHA;
    ys[t] = Xty[b * D_ + t];
    __syncthreads();

    // forward elimination (column-parallel, coalesced row updates)
    for (int k = 0; k < D_ - 1; ++k) {
        colk[t] = (t > k) ? Ab[t * D_ + k] : 0.f;
        __syncthreads();
        const float invp = 1.f / Ab[k * D_ + k];
        const float pr   = Ab[k * D_ + t];
        const float yk   = ys[k];
        for (int i = k + 1; i < D_; ++i)
            Ab[i * D_ + t] -= colk[i] * invp * pr;
        if (t > k) ys[t] -= colk[t] * invp * yk;
        __syncthreads();
    }

    // back substitution
    for (int k = D_ - 1; k >= 0; --k) {
        if (t == 0) xs[k] = ys[k] / Ab[k * D_ + k];
        __syncthreads();
        if (t < k) ys[t] -= Ab[t * D_ + k] * xs[k];
        __syncthreads();
    }
    out[b * D_ + t] = xs[t];
}

// ---------------- host-side launcher ----------------
extern "C" void kernel_launch(void* const* d_in, const int* in_sizes, int n_in,
                              void* d_out, int out_size, void* d_ws, size_t ws_size,
                              hipStream_t stream) {
    const float* x   = (const float*)d_in[0];   // (64,64,64)
    const float* Win = (const float*)d_in[1];   // (1,128)
    const float* Wr1 = (const float*)d_in[2];   // (128,128)
    const float* Wr2 = (const float*)d_in[3];   // (128,128)
    float* out = (float*)d_out;                 // (64,256)

    char* ws = (char*)d_ws;
    const size_t stateBytes = (size_t)HP * WP * CELLH * sizeof(_Float16); // ~66 MB
    const size_t flagsBytes = 256;                                        // 64 ints, padded
    _Float16* state = (_Float16*)ws;
    int*      flags = (int*)(ws + stateBytes);
    _Float16* w1t   = (_Float16*)(ws + stateBytes + flagsBytes);
    _Float16* w2t   = w1t + R_ * R_;
    float*    XtX   = (float*)(ws + stateBytes + flagsBytes
                               + (size_t)2 * R_ * R_ * sizeof(_Float16));
    float*    Xty   = XtX + (size_t)B_ * D_ * D_;

    // zero halo state + progress flags in one pass
    k_zero<<<2048, 256, 0, stream>>>((uint4*)state, (stateBytes + flagsBytes) / 16);
    k_prep_weights<<<64, 256, 0, stream>>>(Wr1, Wr2, w1t, w2t);

    // single persistent launch: 64 column-pipelined workgroups
    k_recur<<<W_, 256, 0, stream>>>(x, Win, w1t, w2t, state, flags);

    k_gram<<<dim3(B_, 4), 256, 0, stream>>>(state, XtX);
    k_xty<<<B_, 256, 0, stream>>>(state, x, Xty);
    k_solve<<<B_, 256, 0, stream>>>(XtX, Xty, out);
}
